// ConditionalRoutedFeedForward_55044300865625
// MI455X (gfx1250) — matrix operand order repaired and checked by reading, not compiled, not run on hardware
//
#include <hip/hip_runtime.h>
#include <math.h>

typedef __attribute__((ext_vector_type(16))) _Float16 v16h;
typedef __attribute__((ext_vector_type(8)))  _Float16 v8h;
typedef __attribute__((ext_vector_type(4)))  _Float16 v4h;
typedef __attribute__((ext_vector_type(8)))  float    v8f;

#define DIM_D   1024
#define NTOK    8192
#define BATCH   4
#define DIM_L   512
#define DIM_H   4096
#define KSEL    1024

// ---------------------------------------------------------------- helpers

__device__ __forceinline__ v8f wmma_f16(v16h a, v16h b, v8f c) {
  // D = A(16x32 f16) * B(32x16 f16) + C(16x16 f32)
  return __builtin_amdgcn_wmma_f32_16x16x32_f16(
      /*neg_a=*/false, a, /*neg_b=*/false, b,
      /*c_mod=*/(short)0, c, /*reuse_a=*/false, /*reuse_b=*/false);
}

__device__ __forceinline__ v8f zero8() {
  v8f z;
#pragma unroll
  for (int i = 0; i < 8; ++i) z[i] = 0.0f;
  return z;
}

__device__ __forceinline__ v16h cat16(v8h lo, v8h hi) {
  v16h r;
#pragma unroll
  for (int i = 0; i < 8; ++i) { r[i] = lo[i]; r[i + 8] = hi[i]; }
  return r;
}

__device__ __forceinline__ float gelu_exact(float v) {
  return 0.5f * v * (1.0f + erff(v * 0.70710678118654752f));
}

// rms_norm one row of 1024 f32 -> f16 into LDS row; optionally also emit
// routing score s[row] = dot(x_row, rt)  (fused to avoid a second pass over x)
template <bool WITH_ROUTE>
__device__ __forceinline__ void load_norm_row(const float* __restrict__ xrow,
                                              const float* __restrict__ gamma,
                                              _Float16* __restrict__ dst, int lane,
                                              const float* __restrict__ rt,
                                              float* __restrict__ s_out) {
  const float4* x4 = (const float4*)xrow;
  const float4* g4 = (const float4*)gamma;
  float4 xr[8];
  float ss = 0.0f, sd = 0.0f;
#pragma unroll
  for (int i = 0; i < 8; ++i) {
    xr[i] = x4[i * 32 + lane];
    ss += xr[i].x * xr[i].x + xr[i].y * xr[i].y + xr[i].z * xr[i].z + xr[i].w * xr[i].w;
    if (WITH_ROUTE) {
      float4 r = ((const float4*)rt)[i * 32 + lane];
      sd += xr[i].x * r.x + xr[i].y * r.y + xr[i].z * r.z + xr[i].w * r.w;
    }
  }
#pragma unroll
  for (int mm = 16; mm >= 1; mm >>= 1) {
    ss += __shfl_xor(ss, mm, 32);
    if (WITH_ROUTE) sd += __shfl_xor(sd, mm, 32);
  }
  if (WITH_ROUTE && lane == 0) *s_out = sd;
  float inv = 32.0f / fmaxf(sqrtf(ss), 1e-12f);  // sqrt(1024)=32
#pragma unroll
  for (int i = 0; i < 8; ++i) {
    float4 g = g4[i * 32 + lane];
    v4h o;
    o[0] = (_Float16)(xr[i].x * inv * g.x);
    o[1] = (_Float16)(xr[i].y * inv * g.y);
    o[2] = (_Float16)(xr[i].z * inv * g.z);
    o[3] = (_Float16)(xr[i].w * inv * g.w);
    *(v4h*)(dst + i * 128 + lane * 4) = o;
  }
}

// ---------------------------------------------------------------- weight pack
// Pack fp32 W[K,N] (row-major) into f16 WMMA-B operand tiles:
// P[((ct*KT + kb)*32 + lane)*16 + e] = W[kb*32 + 8*(lane>>4) + (e&7) + 16*(e>>3)][ct*16 + (lane&15)]
__global__ void pack_b_kernel(const float* __restrict__ W, _Float16* __restrict__ P,
                              int K, int N) {
  int id = blockIdx.x * 256 + threadIdx.x;
  if (id >= K * N) return;
  int e    = id & 15;
  int lane = (id >> 4) & 31;
  int rest = id >> 9;
  int KT   = K >> 5;
  int kb   = rest % KT;
  int ct   = rest / KT;
  int k    = kb * 32 + ((lane >> 4) * 8) + (e & 7) + ((e >> 3) * 16);
  int col  = ct * 16 + (lane & 15);
  P[id] = (_Float16)W[(size_t)k * N + col];
}

// ---------------------------------------------------------------- router + top-k
__device__ __forceinline__ float block_reduce(float v, float* red, int is_max) {
  int lane = threadIdx.x & 31, wid = threadIdx.x >> 5;
#pragma unroll
  for (int m = 16; m >= 1; m >>= 1) {
    float o = __shfl_xor(v, m, 32);
    v = is_max ? fmaxf(v, o) : (v + o);
  }
  if (lane == 0) red[wid] = v;
  __syncthreads();
  if (wid == 0) {
    v = red[lane];
#pragma unroll
    for (int m = 16; m >= 1; m >>= 1) {
      float o = __shfl_xor(v, m, 32);
      v = is_max ? fmaxf(v, o) : (v + o);
    }
    if (lane == 0) red[0] = v;
  }
  __syncthreads();
  float r = red[0];
  __syncthreads();
  return r;
}

__global__ void __launch_bounds__(1024) route_select_kernel(const float* __restrict__ s,
                                                            int* __restrict__ sel) {
  __shared__ unsigned long long keys[NTOK];  // 64 KB; reused as reduction scratch
  float* red = (float*)keys;
  int b   = blockIdx.x;
  int tid = threadIdx.x;
  const float* sb = s + (size_t)b * NTOK;

  float sl[8], bl[8];
#pragma unroll
  for (int i = 0; i < 8; ++i) { sl[i] = sb[tid + 1024 * i]; bl[i] = -sl[i]; }

  const float logk = logf(1152.0f);  // min(1024*9/8, 8192), eps = 1
  float a = 0.0f;
  for (int it = 0; it < 50; ++it) {
    float mx = -3.4e38f;
#pragma unroll
    for (int i = 0; i < 8; ++i) mx = fmaxf(mx, sl[i] + bl[i]);
    mx = block_reduce(mx, red, 1);
    float se = 0.0f;
#pragma unroll
    for (int i = 0; i < 8; ++i) se += expf(sl[i] + bl[i] - mx);
    se = block_reduce(se, red, 0);
    a = logk - (mx + logf(se));
#pragma unroll
    for (int i = 0; i < 8; ++i) bl[i] = -fmaxf(sl[i] + a, 0.0f);
  }
  __syncthreads();

  // score = exp(min(s+a,0)); sort key on min(s+a,0), jax tie-break (lower idx first)
#pragma unroll
  for (int i = 0; i < 8; ++i) {
    int n = tid + 1024 * i;
    float t = fminf(sl[i] + a, 0.0f);
    unsigned u = __float_as_uint(t);
    u = (u & 0x80000000u) ? ~u : (u | 0x80000000u);
    keys[n] = ((unsigned long long)u << 32) | (unsigned)(8191 - n);
  }
  __syncthreads();

  // bitonic sort, descending
  for (unsigned k = 2; k <= 8192u; k <<= 1) {
    for (unsigned j = k >> 1; j > 0; j >>= 1) {
      for (unsigned t = tid; t < 8192u; t += 1024u) {
        unsigned ix = t ^ j;
        if (ix > t) {
          unsigned long long va = keys[t], vb = keys[ix];
          bool desc = ((t & k) == 0);
          if (desc ? (va < vb) : (va > vb)) { keys[t] = vb; keys[ix] = va; }
        }
      }
      __syncthreads();
    }
  }
  sel[b * KSEL + tid] = 8191 - (int)(keys[tid] & 0xFFFFFFFFull);
}

// ---------------------------------------------------------------- light FFN (fused, + routing scores)
__global__ void __launch_bounds__(256) light_ffn_kernel(
    const float* __restrict__ x, const float* __restrict__ rt,
    const float* __restrict__ gamma,
    const _Float16* __restrict__ W1p, const float* __restrict__ b1,
    const _Float16* __restrict__ W2p, const float* __restrict__ b2,
    float* __restrict__ out, float* __restrict__ sbuf) {
  __shared__ _Float16 xn[16 * DIM_D] __attribute__((aligned(16)));  // 32 KB
  __shared__ _Float16 hb[16 * DIM_L] __attribute__((aligned(16)));  // 16 KB
  int lane = threadIdx.x & 31;
  int w    = threadIdx.x >> 5;
  size_t row0 = (size_t)blockIdx.x * 16;

#pragma unroll
  for (int rr = 0; rr < 2; ++rr) {
    int mrow = w * 2 + rr;
    load_norm_row<true>(x + (row0 + mrow) * DIM_D, gamma, xn + mrow * DIM_D, lane,
                        rt, sbuf + row0 + mrow);
  }
  __syncthreads();

  int m = lane & 15, hsel = lane >> 4;

  // phase 1: h = gelu(xn @ W1 + b1), K=1024, each wave owns 4 col-tiles
  {
    v8f acc[4];
    const char* bq[4];
#pragma unroll
    for (int q = 0; q < 4; ++q) {
      acc[q] = zero8();
      bq[q] = (const char*)W1p + (size_t)(w * 4 + q) * 32 * 1024 + lane * 32;  // KT=32
    }
    const _Float16* ap0 = &xn[m * DIM_D + hsel * 8];
#pragma unroll 4
    for (int kb = 0; kb < 32; ++kb) {
      v16h A = cat16(*(const v8h*)(ap0 + kb * 32), *(const v8h*)(ap0 + kb * 32 + 16));
#pragma unroll
      for (int q = 0; q < 4; ++q) {
        v16h B = *(const v16h*)(bq[q] + kb * 1024);
        acc[q] = wmma_f16(A, B, acc[q]);
      }
    }
#pragma unroll
    for (int q = 0; q < 4; ++q) {
      int col = (w * 4 + q) * 16 + m;
      float bias = b1[col];
#pragma unroll
      for (int v = 0; v < 8; ++v) {
        float hv = acc[q][v] + bias;
        hb[(v + 8 * hsel) * DIM_L + col] = (_Float16)gelu_exact(hv);
      }
    }
  }
  __syncthreads();

  // phase 2: out = h @ W2 + b2, K=512, each wave owns 8 col-tiles
  {
    v8f d[8];
    const char* bq[8];
#pragma unroll
    for (int q = 0; q < 8; ++q) {
      d[q] = zero8();
      bq[q] = (const char*)W2p + (size_t)(w * 8 + q) * 16 * 1024 + lane * 32;  // KT=16
    }
    const _Float16* ap0 = &hb[m * DIM_L + hsel * 8];
#pragma unroll 4
    for (int kb = 0; kb < 16; ++kb) {
      v16h A = cat16(*(const v8h*)(ap0 + kb * 32), *(const v8h*)(ap0 + kb * 32 + 16));
#pragma unroll
      for (int q = 0; q < 8; ++q) {
        v16h B = *(const v16h*)(bq[q] + kb * 1024);
        d[q] = wmma_f16(A, B, d[q]);
      }
    }
#pragma unroll
    for (int q = 0; q < 8; ++q) {
      int col = (w * 8 + q) * 16 + m;
      float bias = b2[col];
#pragma unroll
      for (int v = 0; v < 8; ++v)
        out[(row0 + v + 8 * hsel) * DIM_D + col] = d[q][v] + bias;
    }
  }
}

// ---------------------------------------------------------------- heavy FFN part 1
__global__ void __launch_bounds__(256) heavy_ffn1_kernel(
    const float* __restrict__ x, const int* __restrict__ sel,
    const float* __restrict__ gamma, const _Float16* __restrict__ W1p,
    const float* __restrict__ b1, _Float16* __restrict__ H) {
  __shared__ _Float16 xn[16 * DIM_D] __attribute__((aligned(16)));  // 32 KB
  int lane = threadIdx.x & 31;
  int w    = threadIdx.x >> 5;
  int g0   = blockIdx.x * 16;  // flattened selected-token index base

#pragma unroll
  for (int rr = 0; rr < 2; ++rr) {
    int mrow = w * 2 + rr;
    int g = g0 + mrow;
    int b = g >> 10;
    int tok = sel[b * KSEL + (g & 1023)];
    load_norm_row<false>(x + ((size_t)b * NTOK + tok) * DIM_D, gamma,
                         xn + mrow * DIM_D, lane, nullptr, nullptr);
  }
  __syncthreads();

  int m = lane & 15, hsel = lane >> 4;
  const _Float16* ap0 = &xn[m * DIM_D + hsel * 8];
  // 4096 cols -> 256 tiles -> 32 per wave, processed in 8 groups of 4
  for (int grp = 0; grp < 8; ++grp) {
    v8f acc[4];
    const char* bq[4];
#pragma unroll
    for (int q = 0; q < 4; ++q) {
      acc[q] = zero8();
      bq[q] = (const char*)W1p + (size_t)(w * 32 + grp * 4 + q) * 32 * 1024 + lane * 32;  // KT=32
    }
#pragma unroll 4
    for (int kb = 0; kb < 32; ++kb) {
      v16h A = cat16(*(const v8h*)(ap0 + kb * 32), *(const v8h*)(ap0 + kb * 32 + 16));
#pragma unroll
      for (int q = 0; q < 4; ++q) {
        v16h B = *(const v16h*)(bq[q] + kb * 1024);
        acc[q] = wmma_f16(A, B, acc[q]);
      }
    }
#pragma unroll
    for (int q = 0; q < 4; ++q) {
      int col = (w * 32 + grp * 4 + q) * 16 + m;
      float bias = b1[col];
#pragma unroll
      for (int v = 0; v < 8; ++v) {
        float hv = acc[q][v] + bias;
        H[(size_t)(g0 + v + 8 * hsel) * DIM_H + col] = (_Float16)gelu_exact(hv);
      }
    }
  }
}

// ---------------------------------------------------------------- heavy FFN part 2
__global__ void __launch_bounds__(256) heavy_ffn2_kernel(
    const _Float16* __restrict__ H, const int* __restrict__ sel,
    const _Float16* __restrict__ W2p, const float* __restrict__ b2,
    float* __restrict__ out) {
  int lane = threadIdx.x & 31;
  int w    = threadIdx.x >> 5;
  int g0   = blockIdx.x * 16;
  int m = lane & 15, hsel = lane >> 4;

  v8f d[8];
  const char* bq[8];
#pragma unroll
  for (int q = 0; q < 8; ++q) {
    d[q] = zero8();
    bq[q] = (const char*)W2p + (size_t)(w * 8 + q) * 128 * 1024 + lane * 32;  // KT=128
  }
  const _Float16* ap0 = H + (size_t)(g0 + m) * DIM_H + hsel * 8;
#pragma unroll 4
  for (int kb = 0; kb < 128; ++kb) {  // K = 4096
    v16h A = cat16(*(const v8h*)(ap0 + kb * 32), *(const v8h*)(ap0 + kb * 32 + 16));
#pragma unroll
    for (int q = 0; q < 8; ++q) {
      v16h B = *(const v16h*)(bq[q] + kb * 1024);
      d[q] = wmma_f16(A, B, d[q]);
    }
  }
#pragma unroll
  for (int q = 0; q < 8; ++q) {
    int col = (w * 8 + q) * 16 + m;
    float bias = b2[col];
#pragma unroll
    for (int v = 0; v < 8; ++v) {
      int g = g0 + v + 8 * hsel;
      int b = g >> 10;
      int tok = sel[b * KSEL + (g & 1023)];
      size_t o = ((size_t)b * NTOK + tok) * DIM_D + col;
      out[o] += d[q][v] + bias;  // unique (row,col) per lane; light already written
    }
  }
}

// ---------------------------------------------------------------- launch
extern "C" void kernel_launch(void* const* d_in, const int* in_sizes, int n_in,
                              void* d_out, int out_size, void* d_ws, size_t ws_size,
                              hipStream_t stream) {
  (void)in_sizes; (void)n_in; (void)out_size; (void)ws_size;
  const float* x   = (const float*)d_in[0];
  const float* rt  = (const float*)d_in[1];
  const float* lg  = (const float*)d_in[2];
  const float* lw1 = (const float*)d_in[3];
  const float* lb1 = (const float*)d_in[4];
  const float* lw2 = (const float*)d_in[5];
  const float* lb2 = (const float*)d_in[6];
  const float* hg  = (const float*)d_in[7];
  const float* hw1 = (const float*)d_in[8];
  const float* hb1 = (const float*)d_in[9];
  const float* hw2 = (const float*)d_in[10];
  const float* hb2 = (const float*)d_in[11];
  float* out = (float*)d_out;

  char* ws = (char*)d_ws;
  _Float16* W1pL = (_Float16*)(ws);                                    // 1 MiB
  _Float16* W2pL = (_Float16*)(ws + (size_t)(1u  << 20));              // 1 MiB
  _Float16* W1pH = (_Float16*)(ws + (size_t)(2u  << 20));              // 8 MiB
  _Float16* W2pH = (_Float16*)(ws + (size_t)(10u << 20));              // 8 MiB
  float*    sbuf = (float*)   (ws + (size_t)(18u << 20));              // 128 KiB
  int*      sel  = (int*)     (ws + (size_t)(18u << 20) + (1u << 18)); // 16 KiB
  _Float16* H    = (_Float16*)(ws + (size_t)(19u << 20));              // 32 MiB

  pack_b_kernel<<<(DIM_D * DIM_L) / 256, 256, 0, stream>>>(lw1, W1pL, DIM_D, DIM_L);
  pack_b_kernel<<<(DIM_L * DIM_D) / 256, 256, 0, stream>>>(lw2, W2pL, DIM_L, DIM_D);
  pack_b_kernel<<<(DIM_D * DIM_H) / 256, 256, 0, stream>>>(hw1, W1pH, DIM_D, DIM_H);
  pack_b_kernel<<<(DIM_H * DIM_D) / 256, 256, 0, stream>>>(hw2, W2pH, DIM_H, DIM_D);

  // light FFN also produces routing scores (single pass over x)
  light_ffn_kernel<<<(BATCH * NTOK) / 16, 256, 0, stream>>>(x, rt, lg, W1pL, lb1,
                                                            W2pL, lb2, out, sbuf);
  route_select_kernel<<<BATCH, 1024, 0, stream>>>(sbuf, sel);
  heavy_ffn1_kernel<<<(BATCH * KSEL) / 16, 256, 0, stream>>>(x, sel, hg, W1pH, hb1, H);
  heavy_ffn2_kernel<<<(BATCH * KSEL) / 16, 256, 0, stream>>>(H, sel, W2pH, hb2, out);
}
